// Attention_16071767622443
// MI455X (gfx1250) — matrix operand order repaired and checked by reading, not compiled, not run on hardware
//
#include <hip/hip_runtime.h>
#include <hip/hip_bf16.h>

typedef __attribute__((ext_vector_type(16))) _Float16 v16h;
typedef __attribute__((ext_vector_type(8)))  _Float16 v8h;
typedef __attribute__((ext_vector_type(8)))  float    v8f;

#define ATTN_SCALE 0.0625f   // 1/sqrt(256)

union H16U { v16h v; v8h h[2]; };

// Build a 16-half fragment from two 16-byte chunks.
__device__ inline v16h frag2(const _Float16* p0, const _Float16* p1) {
    H16U u;
    u.h[0] = *(const v8h*)p0;
    u.h[1] = *(const v8h*)p1;
    return u.v;
}

// A-matrix fragment (16x32 f16, MxK) from row-major storage (ld in halves).
// ISA layout: lanes 0-15: M=lane, K 0..7 in VGPR0-3, K 16..23 in VGPR4-7;
//             lanes 16-31: M=lane-16, K 8..15 / 24..31.
__device__ inline v16h a_frag(const _Float16* base, int ld, int k0, int lane) {
    int r = lane & 15;
    const _Float16* p = base + r * ld + k0 + ((lane < 16) ? 0 : 8);
    return frag2(p, p + 16);
}

// B-matrix fragment (32x16 f16, KxN) where B[k][n] = src[n0+n][k0+k] (src row-major,
// K contiguous). ISA layout: col = lane&15; lanes 0-15 hold K 0..7 / 16..23, lanes 16-31 K 8..15 / 24..31.
__device__ inline v16h b_frag(const _Float16* src, int ld, int n0, int k0, int lane) {
    int c = lane & 15;
    const _Float16* p = src + (n0 + c) * ld + k0 + ((lane < 16) ? 0 : 8);
    return frag2(p, p + 16);
}

__device__ inline v8f wmma32(v16h a, v16h b, v8f c) {
    return __builtin_amdgcn_wmma_f32_16x16x32_f16(false, a, false, b, (short)0, c,
                                                  false, false);
}

// Async global->LDS 16-byte copy (GV mode): VDST = LDS byte address, VADDR = 64-bit
// global address. Tracked by ASYNCcnt. Generic->LDS address is a truncation on AMDGPU.
// volatile asm => scheduler cannot sink/reorder these against the compute.
__device__ inline void async_ld16(void* lds_dst, const void* gsrc) {
    unsigned int l = (unsigned int)(unsigned long long)lds_dst;
    asm volatile("global_load_async_to_lds_b128 %0, %1, off"
                 :: "v"(l), "v"(gsrc) : "memory");
}
__device__ inline void wait_async0() {
    asm volatile("s_wait_asynccnt 0x0" ::: "memory");
}

// ---------------------------------------------------------------- kernel 0
__global__ __launch_bounds__(256) void cvt_f32_to_f16_kernel(
    const float* __restrict__ s, _Float16* __restrict__ d, int n) {
    int i = blockIdx.x * 256 + threadIdx.x;
    if (i < n) d[i] = (_Float16)s[i];
}

// ---------------------------------------------------------------- kernel 1
// qkv = x @ w_qkv^T + b_qkv ; scatter to Q[B,H,N,32], K[B,H,N,32], Vt[B,H,32,N] (f16)
// Weight tiles (16 rows x 256 = 8 KB) stream through LDS, double-buffered async.
__global__ __launch_bounds__(256) void qkv_proj_kernel(
    const float* __restrict__ x, const _Float16* __restrict__ wqkv,
    const float* __restrict__ bias,
    _Float16* __restrict__ qh, _Float16* __restrict__ kh, _Float16* __restrict__ vth) {
    __shared__ _Float16 xt[128 * 256];          // 64 KB f16 x-tile
    __shared__ _Float16 wt[2][16 * 256];        // 2 x 8 KB weight tiles
    const int lane = threadIdx.x & 31;
    const int wave = threadIdx.x >> 5;
    const int row0 = blockIdx.x * 128;

    // prologue: issue async weight tile 0 (8 KB = 256 threads x 2 b128)
    async_ld16(&wt[0][threadIdx.x * 8], wqkv + threadIdx.x * 8);
    async_ld16(&wt[0][2048 + threadIdx.x * 8], wqkv + 2048 + threadIdx.x * 8);

    // cooperative load + f32->f16 convert: 128x256 floats
    for (int i = threadIdx.x; i < 128 * 256 / 4; i += 256) {
        float4 f = ((const float4*)x)[row0 * 64 + i];
        _Float16* d = &xt[i * 4];
        d[0] = (_Float16)f.x; d[1] = (_Float16)f.y;
        d[2] = (_Float16)f.z; d[3] = (_Float16)f.w;
    }
    __syncthreads();

    const _Float16* arow = &xt[wave * 16 * 256];
    v16h a[8];
#pragma unroll
    for (int kk = 0; kk < 8; ++kk) a[kk] = a_frag(arow, 256, kk * 32, lane);

    for (int t = 0; t < 48; ++t) {              // 48 N-tiles of 16 (768 cols)
        const int cur = t & 1;
        wait_async0();                          // my async writes for tile t done
        __syncthreads();                        // all waves' writes done -> tile ready
        if (t + 1 < 48) {                       // stream tile t+1 into other buffer
            const _Float16* src = wqkv + (size_t)(t + 1) * 16 * 256;
            async_ld16(&wt[cur ^ 1][threadIdx.x * 8], src + threadIdx.x * 8);
            async_ld16(&wt[cur ^ 1][2048 + threadIdx.x * 8], src + 2048 + threadIdx.x * 8);
        }
        const _Float16* wtc = wt[cur];

        v16h bfr[8];
#pragma unroll
        for (int kk = 0; kk < 8; ++kk) bfr[kk] = b_frag(wtc, 256, 0, kk * 32, lane);
        v8f acc = {};
#pragma unroll
        for (int kk = 0; kk < 8; ++kk) acc = wmma32(a[kk], bfr[kk], acc);

        int cc = lane & 15;
        int col = t * 16 + cc;
        float bj = bias[col];
        if (t < 16) {                           // Q columns (uniform branch)
            int h = col >> 5, d = col & 31;
#pragma unroll
            for (int r = 0; r < 8; ++r) {
                int m = row0 + wave * 16 + ((lane < 16) ? r : r + 8);
                int bb = m >> 11, nn = m & 2047;
                qh[(((bb * 8 + h) * 2048) + nn) * 32 + d] = (_Float16)(acc[r] + bj);
            }
        } else if (t < 32) {                    // K columns
            int j = col - 256, h = j >> 5, d = j & 31;
#pragma unroll
            for (int r = 0; r < 8; ++r) {
                int m = row0 + wave * 16 + ((lane < 16) ? r : r + 8);
                int bb = m >> 11, nn = m & 2047;
                kh[(((bb * 8 + h) * 2048) + nn) * 32 + d] = (_Float16)(acc[r] + bj);
            }
        } else {                                // V columns (transposed store)
            int j = col - 512, h = j >> 5, d = j & 31;
#pragma unroll
            for (int r = 0; r < 8; ++r) {
                int m = row0 + wave * 16 + ((lane < 16) ? r : r + 8);
                int bb = m >> 11, nn = m & 2047;
                vth[(((bb * 8 + h) * 32) + d) * 2048 + nn] = (_Float16)(acc[r] + bj);
            }
        }
    }
}

// ---------------------------------------------------------------- kernel 2
// Flash attention: one (b,h) + 128-query tile per block; each wave owns 16 queries.
// K/V tiles stream through LDS via double-buffered async global->LDS loads.
__global__ __launch_bounds__(256) void attn_kernel(
    const _Float16* __restrict__ qh, const _Float16* __restrict__ kh,
    const _Float16* __restrict__ vth, _Float16* __restrict__ zh) {
    __shared__ _Float16 kt[2][64 * 32];         // K tiles  [key][d]   2 x 4 KB
    __shared__ _Float16 vt[2][32 * 64];         // Vt tiles [d][key]   2 x 4 KB
    __shared__ _Float16 ps[8][16 * 64];         // per-wave P scratch  16 KB
    const int lane = threadIdx.x & 31;
    const int wave = threadIdx.x >> 5;
    const int bh = blockIdx.x >> 4;             // b*8+h
    const int q0 = (blockIdx.x & 15) * 128 + wave * 16;
    const _Float16* qbase = qh + (size_t)bh * 2048 * 32;
    const _Float16* kbase = kh + (size_t)bh * 2048 * 32;
    const _Float16* vbase = vth + (size_t)bh * 32 * 2048;

    v16h aq = a_frag(qbase + q0 * 32, 32, 0, lane);   // K-dim = D = 32 exactly
    v8f olo = {}, ohi = {};
    float mrow[8], lrow[8];
#pragma unroll
    for (int r = 0; r < 8; ++r) { mrow[r] = -1e30f; lrow[r] = 0.f; }

    const int vd = threadIdx.x >> 3, vc = threadIdx.x & 7;  // Vt tile mapping

    // prologue: issue async loads for tile 0 (one b128 per thread per tensor)
    async_ld16(&kt[0][threadIdx.x * 8], kbase + threadIdx.x * 8);
    async_ld16(&vt[0][vd * 64 + vc * 8], vbase + (size_t)vd * 2048 + vc * 8);

    for (int it = 0; it < 32; ++it) {
        const int kb = it * 64;
        const int cur = it & 1;
        wait_async0();                          // my async writes done
        __syncthreads();                        // everyone's done -> tile ready
        if (kb + 64 < 2048) {                   // issue next tile into other buffer
            async_ld16(&kt[cur ^ 1][threadIdx.x * 8],
                       kbase + (kb + 64) * 32 + threadIdx.x * 8);
            async_ld16(&vt[cur ^ 1][vd * 64 + vc * 8],
                       vbase + (size_t)vd * 2048 + (kb + 64) + vc * 8);
        }
        const _Float16* ktc = kt[cur];
        const _Float16* vtc = vt[cur];

        // scores: preload 4 B fragments, then 4 WMMAs -> 16x64 f32
        v16h bk[4];
#pragma unroll
        for (int t = 0; t < 4; ++t) bk[t] = b_frag(ktc, 32, t * 16, 0, lane);
        v8f s[4];
#pragma unroll
        for (int t = 0; t < 4; ++t) {
            v8f z = {};
            s[t] = wmma32(aq, bk[t], z);
        }
#pragma unroll
        for (int t = 0; t < 4; ++t)
#pragma unroll
            for (int r = 0; r < 8; ++r) s[t][r] *= ATTN_SCALE;

        // online softmax; row r lives in lanes 0-15 (M=r) / 16-31 (M=r+8)
#pragma unroll
        for (int r = 0; r < 8; ++r) {
            float mx = fmaxf(fmaxf(s[0][r], s[1][r]), fmaxf(s[2][r], s[3][r]));
            for (int off = 8; off >= 1; off >>= 1)
                mx = fmaxf(mx, __shfl_xor(mx, off, 16));
            float mnew = fmaxf(mrow[r], mx);
            float corr = __expf(mrow[r] - mnew);
            float rsum = 0.f;
#pragma unroll
            for (int t = 0; t < 4; ++t) {
                float p = __expf(s[t][r] - mnew);
                s[t][r] = p;
                rsum += p;
            }
            for (int off = 8; off >= 1; off >>= 1)
                rsum += __shfl_xor(rsum, off, 16);
            lrow[r] = lrow[r] * corr + rsum;
            mrow[r] = mnew;
            olo[r] *= corr;
            ohi[r] *= corr;
        }

        // stage P (f16) through per-wave LDS, reload in A layout
        {
            _Float16* pw = ps[wave];
            int cc = lane & 15, rr = (lane < 16) ? 0 : 8;
#pragma unroll
            for (int t = 0; t < 4; ++t)
#pragma unroll
                for (int r = 0; r < 8; ++r)
                    pw[(rr + r) * 64 + t * 16 + cc] = (_Float16)s[t][r];
        }
        asm volatile("s_wait_dscnt 0x0" ::: "memory");  // same-wave LDS RAW
        {
            const _Float16* pw = ps[wave];
            v16h ap0 = a_frag(pw, 64, 0, lane);         // keys 0..31 of tile
            v16h ap1 = a_frag(pw, 64, 32, lane);        // keys 32..63
            v16h bv00 = b_frag(vtc, 64, 0, 0, lane);
            v16h bv01 = b_frag(vtc, 64, 0, 32, lane);
            v16h bv10 = b_frag(vtc, 64, 16, 0, lane);
            v16h bv11 = b_frag(vtc, 64, 16, 32, lane);
            olo = wmma32(ap0, bv00, olo);
            olo = wmma32(ap1, bv01, olo);
            ohi = wmma32(ap0, bv10, ohi);
            ohi = wmma32(ap1, bv11, ohi);
        }
        // no trailing barrier needed: buffer `cur` is only overwritten by the
        // async issue at iteration it+1, which is after that iteration's barrier
    }

    // finalize and write z[b][n][h*32+d] (f16)
    int b = bh >> 3, h = bh & 7, cc = lane & 15;
#pragma unroll
    for (int r = 0; r < 8; ++r) {
        float inv = 1.f / lrow[r];
        float lo = olo[r] * inv, hi = ohi[r] * inv;
        int m = b * 2048 + q0 + ((lane < 16) ? r : r + 8);
        _Float16* zp = zh + (size_t)m * 256 + h * 32;
        zp[cc] = (_Float16)lo;
        zp[16 + cc] = (_Float16)hi;
    }
}

// ---------------------------------------------------------------- kernel 3
// out = z @ w_out^T + b_out  (f32 output); weight tiles double-buffered async in LDS
__global__ __launch_bounds__(256) void out_proj_kernel(
    const _Float16* __restrict__ zh, const _Float16* __restrict__ wout,
    const float* __restrict__ bias, float* __restrict__ out) {
    __shared__ _Float16 zt[128 * 256];          // 64 KB
    __shared__ _Float16 wt[2][16 * 256];        // 2 x 8 KB weight tiles
    const int lane = threadIdx.x & 31;
    const int wave = threadIdx.x >> 5;
    const int row0 = blockIdx.x * 128;

    async_ld16(&wt[0][threadIdx.x * 8], wout + threadIdx.x * 8);
    async_ld16(&wt[0][2048 + threadIdx.x * 8], wout + 2048 + threadIdx.x * 8);

    const uint4* g = (const uint4*)(zh + (size_t)row0 * 256);
    for (int i = threadIdx.x; i < 128 * 256 * 2 / 16; i += 256)
        ((uint4*)zt)[i] = g[i];
    __syncthreads();

    const _Float16* arow = &zt[wave * 16 * 256];
    v16h a[8];
#pragma unroll
    for (int kk = 0; kk < 8; ++kk) a[kk] = a_frag(arow, 256, kk * 32, lane);

    for (int t = 0; t < 16; ++t) {
        const int cur = t & 1;
        wait_async0();
        __syncthreads();
        if (t + 1 < 16) {
            const _Float16* src = wout + (size_t)(t + 1) * 16 * 256;
            async_ld16(&wt[cur ^ 1][threadIdx.x * 8], src + threadIdx.x * 8);
            async_ld16(&wt[cur ^ 1][2048 + threadIdx.x * 8], src + 2048 + threadIdx.x * 8);
        }
        const _Float16* wtc = wt[cur];

        v16h bfr[8];
#pragma unroll
        for (int kk = 0; kk < 8; ++kk) bfr[kk] = b_frag(wtc, 256, 0, kk * 32, lane);
        v8f acc = {};
#pragma unroll
        for (int kk = 0; kk < 8; ++kk) acc = wmma32(a[kk], bfr[kk], acc);

        int col = t * 16 + (lane & 15);
        float bj = bias[col];
#pragma unroll
        for (int r = 0; r < 8; ++r) {
            int m = row0 + wave * 16 + ((lane < 16) ? r : r + 8);
            out[(size_t)m * 256 + col] = acc[r] + bj;
        }
    }
}

// ---------------------------------------------------------------- launch
extern "C" void kernel_launch(void* const* d_in, const int* in_sizes, int n_in,
                              void* d_out, int out_size, void* d_ws, size_t ws_size,
                              hipStream_t stream) {
    const float* x     = (const float*)d_in[0];
    const float* w_qkv = (const float*)d_in[1];
    const float* b_qkv = (const float*)d_in[2];
    const float* w_out = (const float*)d_in[3];
    const float* b_out = (const float*)d_in[4];
    float* out = (float*)d_out;

    char* ws = (char*)d_ws;
    _Float16* wqkv_h = (_Float16*)(ws);                              // 384 KB
    _Float16* wout_h = (_Float16*)(ws + (393216));                   // 128 KB
    _Float16* qh     = (_Float16*)(ws + (524288));                   // 4 MB
    _Float16* kh     = (_Float16*)(ws + (524288 + 4194304));         // 4 MB
    _Float16* vth    = (_Float16*)(ws + (524288 + 2 * 4194304));     // 4 MB
    _Float16* zh     = (_Float16*)(ws + (524288 + 3 * 4194304));     // 4 MB

    cvt_f32_to_f16_kernel<<<(196608 + 255) / 256, 256, 0, stream>>>(w_qkv, wqkv_h, 196608);
    cvt_f32_to_f16_kernel<<<(65536 + 255) / 256, 256, 0, stream>>>(w_out, wout_h, 65536);
    qkv_proj_kernel<<<64, 256, 0, stream>>>(x, wqkv_h, b_qkv, qh, kh, vth);
    attn_kernel<<<4 * 8 * 16, 256, 0, stream>>>(qh, kh, vth, zh);
    out_proj_kernel<<<64, 256, 0, stream>>>(zh, wout_h, b_out, out);
}
